// QAttention_6004364280063
// MI455X (gfx1250) — matrix operand order repaired and checked by reading, not compiled, not run on hardware
//
#include <hip/hip_runtime.h>

// Shapes (fixed by the reference): B=8, N=1024, C=1024, H=16, D=64.
// All matmuls run as bf16 WMMA (v_wmma_f32_16x16x32_bf16) with f32 accumulate.
// LDS staging uses gfx1250 GLOBAL_LOAD_ASYNC_TO_LDS_B128 (ASYNCcnt) with
// double-buffered tiles so the async DMA overlaps the WMMA work.
// Workspace layout (bytes):
//   [0,16M)  x  bf16           [16M,22M) w_qkv bf16    [22M,24M) w_proj bf16
//   [24M,40M) q bf16 [B,H,N,D] [40M,56M) k bf16 [B,H,N,D]
//   [56M,72M) v^T bf16 [B,H,D,N]
//   [72M,88M) attn-out bf16 [B,N,C]
// Requires ws_size >= 88 MiB.

typedef __attribute__((ext_vector_type(16))) __bf16       v16bf;
typedef __attribute__((ext_vector_type(8)))  float        v8f;
typedef __attribute__((ext_vector_type(4)))  unsigned int u32x4;

union FragAB {
  v16bf v;
  u32x4 u[2];
};

__device__ __forceinline__ unsigned short f2bf(float f) {
  unsigned int u = __builtin_bit_cast(unsigned int, f);
  u += 0x7FFFu + ((u >> 16) & 1u);   // round-to-nearest-even
  return (unsigned short)(u >> 16);
}

__device__ __forceinline__ v8f wmma_bf16(const FragAB& a, const FragAB& b, v8f c) {
  // (neg_a, A, neg_b, B, c_mod, C, reuse_a, reuse_b)
  return __builtin_amdgcn_wmma_f32_16x16x32_bf16(false, a.v, false, b.v,
                                                 (short)0, c, false, false);
}

// 16-byte async global->LDS copy (per lane). LDS offset = low 32 bits of the
// generic shared-memory address (aperture mapping keeps addr[31:0] = LDS off).
__device__ __forceinline__ void async_copy16(const unsigned short* lds_dst,
                                             const unsigned short* gsrc) {
  unsigned lds_off = (unsigned)(unsigned long long)lds_dst;
  asm volatile("global_load_async_to_lds_b128 %0, %1, off"
               :: "v"(lds_off), "v"(gsrc)
               : "memory");
}

__device__ __forceinline__ void wait_async0() {
  asm volatile("s_wait_asynccnt 0x0" ::: "memory");
}

// ---------------------------------------------------------------- convert ---
__global__ void cvt_f32_bf16(const float* __restrict__ src,
                             unsigned short* __restrict__ dst, int n) {
  int i = blockIdx.x * blockDim.x + threadIdx.x;
  if (i < n) dst[i] = f2bf(src[i]);
}

// ------------------------------------------------- 128x128 bf16 GEMM core ---
// C[m][n] = sum_k A[m][k] * W[n][k]   (W row-major [N][K], i.e. torch Linear)
// Block: 256 threads = 8 waves; wave (wm,wn) owns a 64x32 region = 4x2 tiles.
// sA/sW are double-buffered: 2 x 128x32 bf16 each (8KB per stage).
__device__ __forceinline__ void gemm128x128_bf16(
    const unsigned short* __restrict__ A,
    const unsigned short* __restrict__ W,
    int K, int mbase, int nbase,
    unsigned short* sA, unsigned short* sW,
    v8f acc[4][2]) {
  const int t    = threadIdx.x;
  const int lane = t & 31;
  const int wave = t >> 5;
  const int wm   = wave & 1;
  const int wn   = wave >> 1;
  const int l15  = lane & 15;
  const int e0   = (lane >> 4) * 8;   // A-frag: lanes 16-31 start at K+8
  const int bo   = (lane >> 4) * 8;   // B-frag: lanes 16-31 start at K+8

  const int lr = t >> 1;              // staging row 0..127
  const int lc = (t & 1) * 16;        // staging col offset (elements)

  const unsigned short* ga = A + (size_t)(mbase + lr) * K + lc;  // + k0
  const unsigned short* gw = W + (size_t)(nbase + lr) * K + lc;
  unsigned short* dA = sA + lr * 32 + lc;   // buffer 1 at +4096 elements
  unsigned short* dW = sW + lr * 32 + lc;

  const v8f vzero = {0.f,0.f,0.f,0.f,0.f,0.f,0.f,0.f};
  for (int i = 0; i < 4; ++i)
    for (int j = 0; j < 2; ++j) acc[i][j] = vzero;

  // prologue: stage k0 = 0 into buffer 0
  async_copy16(dA,     ga);
  async_copy16(dA + 8, ga + 8);
  async_copy16(dW,     gw);
  async_copy16(dW + 8, gw + 8);
  wait_async0();
  __syncthreads();

  for (int k0 = 0; k0 < K; k0 += 32) {
    const int buf = (k0 >> 5) & 1;
    if (k0 + 32 < K) {   // async-prefetch next K-tile into the other buffer
      const int nb = (buf ^ 1) * 4096;
      async_copy16(dA + nb,     ga + k0 + 32);
      async_copy16(dA + nb + 8, ga + k0 + 40);
      async_copy16(dW + nb,     gw + k0 + 32);
      async_copy16(dW + nb + 8, gw + k0 + 40);
    }
    const unsigned short* bA = sA + buf * 4096;
    const unsigned short* bW = sW + buf * 4096;

    FragAB af[4], bw[2];
    for (int i = 0; i < 4; ++i) {
      const unsigned short* base = bA + (wm * 64 + i * 16 + l15) * 32 + e0;
      af[i].u[0] = *(const u32x4*)(base);
      af[i].u[1] = *(const u32x4*)(base + 16);
    }
    for (int j = 0; j < 2; ++j) {
      const unsigned short* base = bW + (wn * 32 + j * 16 + l15) * 32 + bo;
      bw[j].u[0] = *(const u32x4*)(base);
      bw[j].u[1] = *(const u32x4*)(base + 16);
    }
    for (int i = 0; i < 4; ++i)
      for (int j = 0; j < 2; ++j)
        acc[i][j] = wmma_bf16(af[i], bw[j], acc[i][j]);

    wait_async0();       // next-stage async writes landed
    __syncthreads();     // everyone done reading this stage
  }
}

// -------------------------------------------------------------- QKV GEMM ---
__global__ __launch_bounds__(256) void qkv_gemm_kernel(
    const unsigned short* __restrict__ xb,
    const unsigned short* __restrict__ wqkv,
    unsigned short* __restrict__ qws,
    unsigned short* __restrict__ kws,
    unsigned short* __restrict__ vtws) {
  __shared__ unsigned short sA[2 * 128 * 32];
  __shared__ unsigned short sW[2 * 128 * 32];
  const int nbase = blockIdx.x * 128;   // feature tile (0..3071)
  const int mbase = blockIdx.y * 128;   // token tile   (0..8191)
  v8f acc[4][2];
  gemm128x128_bf16(xb, wqkv, 1024, mbase, nbase, sA, sW, acc);

  const int lane = threadIdx.x & 31;
  const int wave = threadIdx.x >> 5;
  const int wm = wave & 1, wn = wave >> 1;
  const int l15 = lane & 15, lh8 = (lane >> 4) * 8;
  for (int i = 0; i < 4; ++i)
    for (int j = 0; j < 2; ++j)
      for (int g = 0; g < 8; ++g) {
        int m = mbase + wm * 64 + i * 16 + g + lh8;
        int f = nbase + wn * 32 + j * 16 + l15;
        int b = m >> 10, n = m & 1023;
        int tsel = f >> 10, hd = f & 1023;
        int h = hd >> 6, d = hd & 63;
        unsigned short val = f2bf(acc[i][j][g]);
        if (tsel == 2) {
          // V stored transposed: [B,H,D,N] so attention can stage V^T rows
          vtws[(((size_t)b * 16 + h) * 64 + d) * 1024 + n] = val;
        } else {
          size_t off = (((size_t)b * 16 + h) * 1024 + n) * 64 + d;
          if (tsel == 0) qws[off] = val;
          else           kws[off] = val;
        }
      }
}

// -------------------------------------------------------- flash attention ---
// grid = (B*H, N/128); block = 256 (8 waves), wave owns 16 query rows.
__global__ __launch_bounds__(256) void attn_kernel(
    const unsigned short* __restrict__ qws,
    const unsigned short* __restrict__ kws,
    const unsigned short* __restrict__ vtws,
    unsigned short* __restrict__ attnb) {
  __shared__ unsigned short sK [2][64 * 64];    // K chunk  [kv][d], dbl-buffer
  __shared__ unsigned short sVt[2][64 * 64];    // V^T chunk [d][kv], dbl-buffer
  __shared__ unsigned short sP [8 * 16 * 64];   // per-wave P staging
  const int bh = blockIdx.x;          // b*16 + h
  const int qt = blockIdx.y;          // query tile of 128
  const int t = threadIdx.x;
  const int lane = t & 31, wave = t >> 5;
  const int l15 = lane & 15;
  const int e0  = (lane >> 4) * 8;
  const int bo  = (lane >> 4) * 8;
  const int lh8 = (lane >> 4) * 8;
  const float scale = 0.125f;         // D^-0.5, D=64

  // Q fragments (16 rows x 64 d) straight from global, A-layout
  FragAB qf[2];
  {
    const int qn = qt * 128 + wave * 16 + l15;
    const unsigned short* qrow = qws + ((size_t)bh * 1024 + qn) * 64;
    for (int kc = 0; kc < 2; ++kc) {
      const unsigned short* p = qrow + kc * 32 + e0;
      qf[kc].u[0] = *(const u32x4*)(p);
      qf[kc].u[1] = *(const u32x4*)(p + 16);
    }
  }

  const v8f vzero = {0.f,0.f,0.f,0.f,0.f,0.f,0.f,0.f};
  v8f o[4];
  for (int dj = 0; dj < 4; ++dj) o[dj] = vzero;
  float rm[8], rl[8];
  for (int g = 0; g < 8; ++g) { rm[g] = -__builtin_inff(); rl[g] = 0.f; }

  const int sr = t >> 2;              // staging row 0..63
  const int sc = (t & 3) * 16;        // staging col (elements)
  // K chunk c: rows c*64+sr of [BH][N][D]; V^T chunk c: cols c*64 of [BH][D][N]
  const unsigned short* kg  = kws  + ((size_t)bh * 1024 + sr) * 64 + sc;  // +c*4096
  const unsigned short* vtg = vtws + ((size_t)bh * 64 + sr) * 1024 + sc;  // +c*64
  unsigned short* dK  = &sK [0][sr * 64 + sc];
  unsigned short* dVt = &sVt[0][sr * 64 + sc];

  // prologue: stage chunk 0 into buffer 0
  async_copy16(dK,      kg);
  async_copy16(dK + 8,  kg + 8);
  async_copy16(dVt,     vtg);
  async_copy16(dVt + 8, vtg + 8);
  wait_async0();
  __syncthreads();

  for (int c = 0; c < 16; ++c) {
    const int buf = c & 1;
    if (c + 1 < 16) {  // async-prefetch next KV chunk into the other buffer
      const int nb = (buf ^ 1) * 4096;
      const unsigned short* kn  = kg  + (size_t)(c + 1) * 4096;
      const unsigned short* vtn = vtg + (size_t)(c + 1) * 64;
      async_copy16(dK + nb,      kn);
      async_copy16(dK + nb + 8,  kn + 8);
      async_copy16(dVt + nb,     vtn);
      async_copy16(dVt + nb + 8, vtn + 8);
    }
    const unsigned short* bK  = sK [buf];
    const unsigned short* bVt = sVt[buf];

    // S = Q @ K^T  (16 x 64 per wave, 4 tiles x 2 K-steps)
    v8f s[4];
    for (int j = 0; j < 4; ++j) {
      s[j] = vzero;
      for (int kc = 0; kc < 2; ++kc) {
        FragAB kf;
        const unsigned short* base = bK + (j * 16 + l15) * 64 + kc * 32 + bo;
        kf.u[0] = *(const u32x4*)(base);
        kf.u[1] = *(const u32x4*)(base + 16);
        s[j] = wmma_bf16(qf[kc], kf, s[j]);
      }
    }

    // online softmax; row stats reduced over 16-lane halves
    for (int g = 0; g < 8; ++g) {
      float mx = -__builtin_inff();
      for (int j = 0; j < 4; ++j) { s[j][g] *= scale; mx = fmaxf(mx, s[j][g]); }
      for (int off = 8; off >= 1; off >>= 1) mx = fmaxf(mx, __shfl_xor(mx, off, 32));
      float nm = fmaxf(rm[g], mx);
      float corr = __expf(rm[g] - nm);
      rm[g] = nm;
      float rs = 0.f;
      for (int j = 0; j < 4; ++j) {
        float p = __expf(s[j][g] - nm);
        s[j][g] = p;
        rs += p;
      }
      for (int off = 8; off >= 1; off >>= 1) rs += __shfl_xor(rs, off, 32);
      rl[g] = rl[g] * corr + rs;
      for (int dj = 0; dj < 4; ++dj) o[dj][g] *= corr;
    }

    // re-layout P (C-layout -> A-layout) through per-wave LDS region, as bf16.
    // Region is wave-private; DS ops are in-order within a wave, no barrier.
    unsigned short* myP = sP + wave * 16 * 64;
    for (int j = 0; j < 4; ++j)
      for (int g = 0; g < 8; ++g)
        myP[(g + lh8) * 64 + j * 16 + l15] = f2bf(s[j][g]);

    // O += P @ V   (B operand from V^T rows)
    for (int kc = 0; kc < 2; ++kc) {
      FragAB pf;
      const unsigned short* pb = myP + l15 * 64 + kc * 32 + e0;
      pf.u[0] = *(const u32x4*)(pb);
      pf.u[1] = *(const u32x4*)(pb + 16);
      for (int dj = 0; dj < 4; ++dj) {
        FragAB vf;
        const unsigned short* vb = bVt + (dj * 16 + l15) * 64 + kc * 32 + bo;
        vf.u[0] = *(const u32x4*)(vb);
        vf.u[1] = *(const u32x4*)(vb + 16);
        o[dj] = wmma_bf16(pf, vf, o[dj]);
      }
    }

    wait_async0();     // next chunk's async writes landed
    __syncthreads();   // everyone done reading this chunk
  }

  // normalize and store to [B, N, H*D] bf16
  const int b = bh >> 4, h = bh & 15;
  for (int dj = 0; dj < 4; ++dj)
    for (int g = 0; g < 8; ++g) {
      int qn = qt * 128 + wave * 16 + g + lh8;
      float val = o[dj][g] / rl[g];
      attnb[((size_t)b * 1024 + qn) * 1024 + h * 64 + dj * 16 + l15] = f2bf(val);
    }
}

// ------------------------------------------------------------- out proj ---
__global__ __launch_bounds__(256) void proj_gemm_kernel(
    const unsigned short* __restrict__ attnb,
    const unsigned short* __restrict__ wproj,
    const float* __restrict__ bias,
    float* __restrict__ out) {
  __shared__ unsigned short sA[2 * 128 * 32];
  __shared__ unsigned short sW[2 * 128 * 32];
  const int nbase = blockIdx.x * 128;
  const int mbase = blockIdx.y * 128;
  v8f acc[4][2];
  gemm128x128_bf16(attnb, wproj, 1024, mbase, nbase, sA, sW, acc);

  const int lane = threadIdx.x & 31;
  const int wave = threadIdx.x >> 5;
  const int wm = wave & 1, wn = wave >> 1;
  const int l15 = lane & 15, lh8 = (lane >> 4) * 8;
  for (int i = 0; i < 4; ++i)
    for (int j = 0; j < 2; ++j)
      for (int g = 0; g < 8; ++g) {
        int m = mbase + wm * 64 + i * 16 + g + lh8;
        int f = nbase + wn * 32 + j * 16 + l15;
        out[(size_t)m * 1024 + f] = acc[i][j][g] + bias[f];
      }
}

// ---------------------------------------------------------------- launch ---
extern "C" void kernel_launch(void* const* d_in, const int* in_sizes, int n_in,
                              void* d_out, int out_size, void* d_ws, size_t ws_size,
                              hipStream_t stream) {
  const float* x      = (const float*)d_in[0];
  const float* w_qkv  = (const float*)d_in[1];
  const float* w_proj = (const float*)d_in[2];
  const float* b_proj = (const float*)d_in[3];
  float* out = (float*)d_out;

  char* ws = (char*)d_ws;
  unsigned short* xb     = (unsigned short*)(ws);
  unsigned short* wqkvb  = (unsigned short*)(ws + (16ull << 20));
  unsigned short* wprojb = (unsigned short*)(ws + (22ull << 20));
  unsigned short* qws    = (unsigned short*)(ws + (24ull << 20));
  unsigned short* kws    = (unsigned short*)(ws + (40ull << 20));
  unsigned short* vtws   = (unsigned short*)(ws + (56ull << 20));
  unsigned short* attnb  = (unsigned short*)(ws + (72ull << 20));

  const int nx  = 8 * 1024 * 1024;
  const int nw1 = 3 * 1024 * 1024;
  const int nw2 = 1024 * 1024;
  cvt_f32_bf16<<<(nx  + 255) / 256, 256, 0, stream>>>(x,      xb,     nx);
  cvt_f32_bf16<<<(nw1 + 255) / 256, 256, 0, stream>>>(w_qkv,  wqkvb,  nw1);
  cvt_f32_bf16<<<(nw2 + 255) / 256, 256, 0, stream>>>(w_proj, wprojb, nw2);

  qkv_gemm_kernel <<<dim3(24, 64), 256, 0, stream>>>(xb, wqkvb, qws, kws, vtws);
  attn_kernel     <<<dim3(128, 8), 256, 0, stream>>>(qws, kws, vtws, attnb);
  proj_gemm_kernel<<<dim3(8, 64),  256, 0, stream>>>(attnb, wprojb, b_proj, out);
}
